// HyperbolicTransformerEncoder_11390253269607
// MI455X (gfx1250) — matrix operand order repaired and checked by reading, not compiled, not run on hardware
//
#include <hip/hip_runtime.h>

// ---------------------------------------------------------------------------
// Hyperbolic transformer encoder for MI455X (gfx1250), wave32 + WMMA bf16.
// - All GEMM-shaped work: v_wmma_f32_16x16x32_bf16, f32 accumulate.
// - GEMM register-blocked 64x32 per wave: 8 WMMA per 12 b128 loads.
// - c=1 identity: exp(-2*atanh(a)) == (1-a)/(1+a); with a = sqrt(num)/|den|
//   this is (|den|-sqrt(num))/(|den|+sqrt(num)) -> 1 sqrt + 1 rcp per element.
// - Working set ~30MB -> L2-resident (192MB); no HBM pressure, compute bound.
// ---------------------------------------------------------------------------

typedef unsigned short u16;
typedef __attribute__((ext_vector_type(16))) __bf16 v16bf;
typedef __attribute__((ext_vector_type(8)))  float  v8f;
typedef __attribute__((ext_vector_type(4)))  float  v4f;
typedef __attribute__((ext_vector_type(8)))  u16    v8u16;
typedef __attribute__((ext_vector_type(4)))  u16    v4u16;

union FragBF16 { v16bf bf; v8u16 h[2]; };

#define DMODEL 512
#define NHEAD  8
#define HDIM   64
#define SEQLEN 1024
#define NTOK   4096   // BATCH * SEQ

__device__ __forceinline__ u16 f2bf(float f) {
  unsigned u = __builtin_bit_cast(unsigned, f);
  u += 0x7FFFu + ((u >> 16) & 1u);           // round-to-nearest-even
  return (u16)(u >> 16);
}
__device__ __forceinline__ float bf2f(u16 h) {
  unsigned u = ((unsigned)h) << 16;
  return __builtin_bit_cast(float, u);
}

__device__ __forceinline__ v8f wmma_bf16(const FragBF16& a, const FragBF16& b, v8f c) {
  return __builtin_amdgcn_wmma_f32_16x16x32_bf16(false, a.bf, false, b.bf,
                                                 (short)0, c, false, false);
}

// --------------------------- weight f32 -> bf16 ----------------------------
__global__ __launch_bounds__(128) void k_cvt(const float* __restrict__ src,
                                             u16* __restrict__ dst, int n) {
  int i = (blockIdx.x * blockDim.x + threadIdx.x) * 4;
  if (i >= n) return;
  v4f v = *(const v4f*)(src + i);
  v4u16 o;
  o[0] = f2bf(v[0]); o[1] = f2bf(v[1]); o[2] = f2bf(v[2]); o[3] = f2bf(v[3]);
  *(v4u16*)(dst + i) = o;
}

// ------------------- embedding gather + positional add ---------------------
__global__ __launch_bounds__(128) void k_embed(const int* __restrict__ idx,
                                               const float* __restrict__ emb,
                                               const float* __restrict__ pos,
                                               u16* __restrict__ xb) {
  int t = blockIdx.x;                 // token 0..4095
  int j = threadIdx.x * 4;            // 128 threads x 4 = 512 dims
  int id = idx[t];
  int s  = t & (SEQLEN - 1);
  v4f e = *(const v4f*)(emb + (long)id * DMODEL + j);
  v4f p = *(const v4f*)(pos + (long)s  * DMODEL + j);
  v4u16 o;
  o[0] = f2bf(e[0] + p[0]); o[1] = f2bf(e[1] + p[1]);
  o[2] = f2bf(e[2] + p[2]); o[3] = f2bf(e[3] + p[3]);
  *(v4u16*)(xb + (long)t * DMODEL + j) = o;
}

// ------------------------- per-row squared norms ---------------------------
__global__ __launch_bounds__(256) void k_norms(const u16* __restrict__ Qb,
                                               const u16* __restrict__ Kb,
                                               float* __restrict__ qn,
                                               float* __restrict__ kn) {
  int r = blockIdx.x * blockDim.x + threadIdx.x;   // 0..32767  (bh*1024 + s)
  const v8u16* q = (const v8u16*)(Qb + (long)r * HDIM);
  const v8u16* k = (const v8u16*)(Kb + (long)r * HDIM);
  float sq = 0.f, sk = 0.f;
#pragma unroll
  for (int i = 0; i < 8; ++i) {
    v8u16 cq = q[i], ck = k[i];
#pragma unroll
    for (int j = 0; j < 8; ++j) {
      float a = bf2f(cq[j]); sq = __builtin_fmaf(a, a, sq);
      float b = bf2f(ck[j]); sk = __builtin_fmaf(b, b, sk);
    }
  }
  qn[r] = sq; kn[r] = sk;
}

// ------------------------------ WMMA GEMM ----------------------------------
// C[4096,512] = A[4096,512] @ W[512,512]^T + bias.
// Register blocking: each wave computes a 64x32 tile (4 M-subtiles x 2 N-
// subtiles, 8 f32 accumulators). Per K=32 step: 8 A-loads + 4 B-loads (b128)
// feed 8 WMMAs -> A fragments reused x2, B fragments reused x4.
// MODE 0: bf16 out at [(b*8+h)*1024+s]*64+d          (Q,K layout)
// MODE 1: bf16 out at [(b*8+h)*64+d]*1024+s          (V transposed layout)
// MODE 2: bf16 out at m*512+n                        (attention-out / hidden)
// MODE 3: f32 relu out at m*512+n                    (final output)
template <int MODE>
__global__ __launch_bounds__(256) void k_gemm(const u16* __restrict__ A,
                                              const u16* __restrict__ W,
                                              const float* __restrict__ bias,
                                              void* __restrict__ out) {
  int wid  = blockIdx.x * 8 + (threadIdx.x >> 5);   // 1024 waves total
  int lane = threadIdx.x & 31;
  int l15  = lane & 15, hi = lane >> 4;
  int m0 = (wid >> 4) * 64;                          // 64 M-blocks (64 rows)
  int n0 = (wid & 15) * 32;                          // 16 N-blocks (32 cols)

  const u16* arow0 = A + (long)(m0 + l15) * DMODEL;  // A rows, contiguous in K
  const u16* wrow0 = W + (long)(n0 + l15) * DMODEL;  // B col n == W row n
  v8f acc[4][2] = {};
  for (int kk = 0; kk < DMODEL; kk += 32) {
    FragBF16 b[2];
#pragma unroll
    for (int nb = 0; nb < 2; ++nb) {
      const u16* wr = wrow0 + (long)(nb * 16) * DMODEL;
      b[nb].h[0] = *(const v8u16*)(wr + kk + 8 * hi);        // B: K {0..7|8..15}
      b[nb].h[1] = *(const v8u16*)(wr + kk + 16 + 8 * hi);
    }
#pragma unroll
    for (int ma = 0; ma < 4; ++ma) {
      const u16* ar = arow0 + (long)(ma * 16) * DMODEL;
      FragBF16 a;
      a.h[0] = *(const v8u16*)(ar + kk + 8 * hi);            // A: K {0..7|8..15}
      a.h[1] = *(const v8u16*)(ar + kk + 16 + 8 * hi);       //    K {16..23|24..31}
      acc[ma][0] = wmma_bf16(a, b[0], acc[ma][0]);
      acc[ma][1] = wmma_bf16(a, b[1], acc[ma][1]);
    }
  }

#pragma unroll
  for (int nb = 0; nb < 2; ++nb) {
    int n  = n0 + nb * 16 + l15;
    float bv = bias[n];
#pragma unroll
    for (int ma = 0; ma < 4; ++ma) {
      int mrow  = m0 + ma * 16;
      if constexpr (MODE == 0) {
        int h = n >> 6, d = n & 63;
        int bb = mrow >> 10, sbase = (mrow & (SEQLEN - 1)) + 8 * hi;
        u16* ob = (u16*)out + ((long)(bb * NHEAD + h) * SEQLEN) * HDIM + d;
#pragma unroll
        for (int v = 0; v < 8; ++v)
          ob[(long)(sbase + v) * HDIM] = f2bf(acc[ma][nb][v] + bv);
      } else if constexpr (MODE == 1) {
        int h = n >> 6, d = n & 63;
        int bb = mrow >> 10, sbase = (mrow & (SEQLEN - 1)) + 8 * hi;
        v8u16 o;
#pragma unroll
        for (int v = 0; v < 8; ++v) o[v] = f2bf(acc[ma][nb][v] + bv);
        *(v8u16*)((u16*)out + ((long)(bb * NHEAD + h) * HDIM + d) * SEQLEN + sbase) = o;
      } else if constexpr (MODE == 2) {
        u16* ob = (u16*)out + (long)(mrow + 8 * hi) * DMODEL + n;
#pragma unroll
        for (int v = 0; v < 8; ++v) ob[(long)v * DMODEL] = f2bf(acc[ma][nb][v] + bv);
      } else {
        float* ob = (float*)out + (long)(mrow + 8 * hi) * DMODEL + n;
#pragma unroll
        for (int v = 0; v < 8; ++v)
          ob[(long)v * DMODEL] = fmaxf(acc[ma][nb][v] + bv, 0.f);
      }
    }
  }
}

// --------------------------- hyperbolic attention --------------------------
// One wave handles one (b,h, 16-query tile); streams keys 32 at a time.
// w in (0,1] -> single-pass accumulation, no running-max rescale needed.
__global__ __launch_bounds__(256) void k_attn(const u16* __restrict__ Qb,
                                              const u16* __restrict__ Kb,
                                              const u16* __restrict__ Vt,
                                              const float* __restrict__ qn,
                                              const float* __restrict__ kn,
                                              u16* __restrict__ attn) {
  __shared__ u16 lds[8 * 16 * 32];                    // 1 KB relayout tile/wave
  int wib  = threadIdx.x >> 5;
  int wid  = blockIdx.x * 8 + wib;                    // 2048 tasks
  int lane = threadIdx.x & 31;
  int l15  = lane & 15, hi = lane >> 4;
  int bh = wid >> 6, q0 = (wid & 63) * 16;

  const u16* Q = Qb + (long)bh * SEQLEN * HDIM;
  const u16* K = Kb + (long)bh * SEQLEN * HDIM;
  const u16* V = Vt + (long)bh * HDIM * SEQLEN;
  const float* qnp = qn + (long)bh * SEQLEN;
  const float* knp = kn + (long)bh * SEQLEN;
  u16* ldst = lds + wib * 512;

  // Q A-fragments (16x64 = two K=32 fragments), resident for the whole loop
  FragBF16 aq0, aq1;
  const u16* qrow = Q + (long)(q0 + l15) * HDIM;
  aq0.h[0] = *(const v8u16*)(qrow + 0  + 8 * hi);
  aq0.h[1] = *(const v8u16*)(qrow + 16 + 8 * hi);
  aq1.h[0] = *(const v8u16*)(qrow + 32 + 8 * hi);
  aq1.h[1] = *(const v8u16*)(qrow + 48 + 8 * hi);

  float qnr[8];                                        // |q|^2 for rows v+8*hi
  {
    v4f t0 = *(const v4f*)(qnp + q0 + 8 * hi);
    v4f t1 = *(const v4f*)(qnp + q0 + 8 * hi + 4);
    qnr[0] = t0[0]; qnr[1] = t0[1]; qnr[2] = t0[2]; qnr[3] = t0[3];
    qnr[4] = t1[0]; qnr[5] = t1[1]; qnr[6] = t1[2]; qnr[7] = t1[3];
  }

  v8f o0 = {}, o1 = {}, o2 = {}, o3 = {};
  float rs[8] = {0.f, 0.f, 0.f, 0.f, 0.f, 0.f, 0.f, 0.f};
  const float ONE_M_EPS = 1.f - 1e-7f;

  for (int kb = 0; kb < SEQLEN; kb += 32) {
#pragma unroll
    for (int nt = 0; nt < 2; ++nt) {                   // two 16-key column tiles
      int key0 = kb + nt * 16;
      const u16* krow = K + (long)(key0 + l15) * HDIM; // B col n == key row n
      FragBF16 b0, b1;
      b0.h[0] = *(const v8u16*)(krow + 8 * hi);
      b0.h[1] = *(const v8u16*)(krow + 16 + 8 * hi);
      b1.h[0] = *(const v8u16*)(krow + 32 + 8 * hi);
      b1.h[1] = *(const v8u16*)(krow + 48 + 8 * hi);
      v8f s = {};
      s = wmma_bf16(aq0, b0, s);                       // S = Q K^T  (K-dim 64)
      s = wmma_bf16(aq1, b1, s);
      float knv = knp[key0 + l15];
#pragma unroll
      for (int v = 0; v < 8; ++v) {                    // Poincare dist -> w
        float uy  = -s[v];
        float qv  = qnr[v];
        float t   = __builtin_fmaf(2.f, uy, 1.f);      // 1 + 2*uy
        float Am  = t + knv;
        float Bm  = 1.f - qv;
        float AB2 = (Am * Bm) * (t - 1.f);             // 2*A*B*uy
        float num = __builtin_fmaf(Am * Am, qv,
                    __builtin_fmaf(Bm * Bm, knv, AB2));
        float den = fabsf(__builtin_fmaf(qv, knv, t)); // |den|; m = sqrt(num)/|den|
        float r   = __builtin_amdgcn_sqrtf(fmaxf(num, 0.f));
        r         = fminf(r, ONE_M_EPS * den);         // clip arg <= 1-eps
        // w = exp(-2*atanh(r/den)) = (den - r)/(den + r)
        float w   = (den - r) * __builtin_amdgcn_rcpf(den + r);
        rs[v] += w;
        ldst[(v + 8 * hi) * 32 + nt * 16 + l15] = f2bf(w);   // C -> A relayout
      }
    }
    asm volatile("s_wait_dscnt 0x0" ::: "memory");     // DS in-order per wave
    FragBF16 aw;                                       // w as A-fragment 16x32
    aw.h[0] = *(const v8u16*)(ldst + l15 * 32 + 8 * hi);
    aw.h[1] = *(const v8u16*)(ldst + l15 * 32 + 16 + 8 * hi);
#pragma unroll
    for (int dt = 0; dt < 4; ++dt) {                   // out += w @ V  (N=64)
      const u16* vcol = V + (long)(dt * 16 + l15) * SEQLEN + kb + 8 * hi;
      FragBF16 bv;
      bv.h[0] = *(const v8u16*)(vcol);                 // V^T: keys {0..7|8..15}
      bv.h[1] = *(const v8u16*)(vcol + 16);
      v8f& oc = (dt == 0) ? o0 : (dt == 1) ? o1 : (dt == 2) ? o2 : o3;
      oc = wmma_bf16(aw, bv, oc);
    }
  }

#pragma unroll
  for (int v = 0; v < 8; ++v) {                        // row-sum across lane half
    float r = rs[v];
    r += __shfl_xor(r, 1, 32);
    r += __shfl_xor(r, 2, 32);
    r += __shfl_xor(r, 4, 32);
    r += __shfl_xor(r, 8, 32);
    rs[v] = __builtin_amdgcn_rcpf(r);
  }
  int bb = bh >> 3, h = bh & 7;
  u16* obase = attn + (long)bb * SEQLEN * DMODEL + h * HDIM;
#pragma unroll
  for (int v = 0; v < 8; ++v) {
    u16* orow = obase + (long)(q0 + v + 8 * hi) * DMODEL;
    orow[ 0 + l15] = f2bf(o0[v] * rs[v]);
    orow[16 + l15] = f2bf(o1[v] * rs[v]);
    orow[32 + l15] = f2bf(o2[v] * rs[v]);
    orow[48 + l15] = f2bf(o3[v] * rs[v]);
  }
}

// ------------------------------- launcher ----------------------------------
extern "C" void kernel_launch(void* const* d_in, const int* in_sizes, int n_in,
                              void* d_out, int out_size, void* d_ws, size_t ws_size,
                              hipStream_t stream) {
  (void)in_sizes; (void)n_in; (void)out_size; (void)ws_size;
  const int*   indices = (const int*)  d_in[0];
  const float* emb     = (const float*)d_in[1];
  const float* pos     = (const float*)d_in[2];
  const float* Wq = (const float*)d_in[3];  const float* bq = (const float*)d_in[4];
  const float* Wk = (const float*)d_in[5];  const float* bk = (const float*)d_in[6];
  const float* Wv = (const float*)d_in[7];  const float* bv = (const float*)d_in[8];
  const float* Wo = (const float*)d_in[9];  const float* bo = (const float*)d_in[10];
  const float* Wf = (const float*)d_in[11]; const float* bf = (const float*)d_in[12];
  float* out = (float*)d_out;

  char* ws = (char*)d_ws;
  size_t off = 0;
  auto alloc = [&](size_t bytes) -> void* {
    void* p = ws + off;
    off = (off + bytes + 255) & ~(size_t)255;
    return p;
  };
  u16* xb   = (u16*)alloc((size_t)NTOK * DMODEL * 2);          // bf16 activations
  u16* Qb   = (u16*)alloc((size_t)NTOK * DMODEL * 2);          // [bh][s][64]
  u16* Kb   = (u16*)alloc((size_t)NTOK * DMODEL * 2);          // [bh][s][64]
  u16* Vt   = (u16*)alloc((size_t)NTOK * DMODEL * 2);          // [bh][64][s]
  u16* at   = (u16*)alloc((size_t)NTOK * DMODEL * 2);          // attention out
  u16* h1   = (u16*)alloc((size_t)NTOK * DMODEL * 2);          // after Wo
  float* qn = (float*)alloc((size_t)32 * SEQLEN * 4);
  float* kn = (float*)alloc((size_t)32 * SEQLEN * 4);
  u16* Wqb = (u16*)alloc((size_t)DMODEL * DMODEL * 2);
  u16* Wkb = (u16*)alloc((size_t)DMODEL * DMODEL * 2);
  u16* Wvb = (u16*)alloc((size_t)DMODEL * DMODEL * 2);
  u16* Wob = (u16*)alloc((size_t)DMODEL * DMODEL * 2);
  u16* Wfb = (u16*)alloc((size_t)DMODEL * DMODEL * 2);

  const int NW = DMODEL * DMODEL;                              // 262144
  k_cvt<<<512, 128, 0, stream>>>(Wq, Wqb, NW);
  k_cvt<<<512, 128, 0, stream>>>(Wk, Wkb, NW);
  k_cvt<<<512, 128, 0, stream>>>(Wv, Wvb, NW);
  k_cvt<<<512, 128, 0, stream>>>(Wo, Wob, NW);
  k_cvt<<<512, 128, 0, stream>>>(Wf, Wfb, NW);

  k_embed<<<NTOK, 128, 0, stream>>>(indices, emb, pos, xb);

  k_gemm<0><<<128, 256, 0, stream>>>(xb, Wqb, bq, (void*)Qb); // Q
  k_gemm<0><<<128, 256, 0, stream>>>(xb, Wkb, bk, (void*)Kb); // K
  k_gemm<1><<<128, 256, 0, stream>>>(xb, Wvb, bv, (void*)Vt); // V (transposed)

  k_norms<<<128, 256, 0, stream>>>(Qb, Kb, qn, kn);

  k_attn<<<256, 256, 0, stream>>>(Qb, Kb, Vt, qn, kn, at);

  k_gemm<2><<<128, 256, 0, stream>>>(at, Wob, bo, (void*)h1); // @Wo^T + bo
  k_gemm<3><<<128, 256, 0, stream>>>(h1, Wfb, bf, (void*)out);// @Wf^T + bf, relu
}
